// SparseLinear_64312840290398
// MI455X (gfx1250) — matrix operand order repaired and checked by reading, compiled, not run on hardware
//
#include <hip/hip_runtime.h>

typedef __attribute__((ext_vector_type(16))) _Float16 v16h;
typedef __attribute__((ext_vector_type(8)))  _Float16 v8h;
typedef __attribute__((ext_vector_type(8)))  float    v8f;
typedef __attribute__((ext_vector_type(4)))  float    v4f;
typedef __attribute__((ext_vector_type(4)))  int      v4i;

typedef __attribute__((address_space(1))) v4i gv4i;   // global-AS 16B chunk
typedef __attribute__((address_space(3))) v4i sv4i;   // LDS-AS 16B chunk

constexpr int BATCH = 4096;   // tokens
constexpr int MDIM  = 4096;   // reduction dim (K)
constexpr int NDIM  = 4096;   // output features

// ---------------------------------------------------------------------------
// async global->LDS copy (GLOBAL_LOAD_ASYNC_TO_LDS_B128, ASYNCcnt-tracked)
// signature (from clang diagnostic): (v4i AS1*, v4i AS3*, int offset, int cpol)
// ---------------------------------------------------------------------------
__device__ __forceinline__ void async_copy_b128(const _Float16* g, _Float16* l)
{
#if __has_builtin(__builtin_amdgcn_global_load_async_to_lds_b128)
    __builtin_amdgcn_global_load_async_to_lds_b128(
        (gv4i*)g, (sv4i*)l, /*offset=*/0, /*cpol=*/0);
#else
    unsigned l32 = (unsigned)(uintptr_t)l;   // LDS byte offset (addr[31:0])
    asm volatile("global_load_async_to_lds_b128 %0, %1, off"
                 :: "v"(l32), "v"(g) : "memory");
#endif
}

__device__ __forceinline__ void wait_asynccnt0()
{
#if __has_builtin(__builtin_amdgcn_s_wait_asynccnt)
    __builtin_amdgcn_s_wait_asynccnt(0);
#else
    asm volatile("s_wait_asynccnt 0" ::: "memory");
#endif
}

// ---------------------------------------------------------------------------
// 1) x (f32) -> xh (f16), 8 elements per thread, b128 in / b128 out
// ---------------------------------------------------------------------------
__global__ void cvt_f32_to_f16(const float* __restrict__ src,
                               _Float16* __restrict__ dst, long n)
{
    long stride = (long)gridDim.x * blockDim.x;
    for (long i = (long)blockIdx.x * blockDim.x + threadIdx.x; i * 8 < n; i += stride) {
        long e = i * 8;
        v4f a = *(const v4f*)(src + e);
        v4f b = *(const v4f*)(src + e + 4);
        v8h o = { (_Float16)a.x, (_Float16)a.y, (_Float16)a.z, (_Float16)a.w,
                  (_Float16)b.x, (_Float16)b.y, (_Float16)b.z, (_Float16)b.w };
        *(v8h*)(dst + e) = o;
    }
}

// ---------------------------------------------------------------------------
// 2) zero the dense f16 weight buffer
// ---------------------------------------------------------------------------
__global__ void zero_f16(_Float16* __restrict__ dst, long n)
{
    long stride = (long)gridDim.x * blockDim.x;
    v8h z = {};
    for (long i = (long)blockIdx.x * blockDim.x + threadIdx.x; i * 8 < n; i += stride)
        *(v8h*)(dst + i * 8) = z;
}

// ---------------------------------------------------------------------------
// 3) CSR scatter: Wd[row][col] = (f16)val   (dense row-major [NDIM][MDIM])
// ---------------------------------------------------------------------------
__global__ void scatter_csr(const float* __restrict__ val,
                            const int* __restrict__ rows,
                            const int* __restrict__ cols,
                            _Float16* __restrict__ wd, int nnz)
{
    int i = blockIdx.x * blockDim.x + threadIdx.x;
    if (i < nnz)
        wd[(long)rows[i] * MDIM + cols[i]] = (_Float16)val[i];
}

// ---------------------------------------------------------------------------
// 4) WMMA GEMM with double-buffered async-LDS staging.
//    block = 256 threads = 8 wave32; block tile 128(t) x 256(n)
//    wave tile 64x64 => 4x4 grid of v_wmma_f32_16x16x32_f16 per K-step of 32.
//    LDS rows padded to 40 halves (80 B) => conflict-free ds_load_b128.
// ---------------------------------------------------------------------------
constexpr int RS     = 40;            // LDS row stride in halves (32 + 8 pad)
constexpr int ABUF   = 128 * RS;      // halves per A stage
constexpr int BBUF   = 256 * RS;      // halves per B stage

__global__ void __launch_bounds__(256)
gemm_wmma_f16(const _Float16* __restrict__ xh,
              const _Float16* __restrict__ wd,
              const float*    __restrict__ bias,
              float*          __restrict__ out)
{
    __shared__ _Float16 As[2 * ABUF];  // 20 KB
    __shared__ _Float16 Bs[2 * BBUF];  // 40 KB

    const int tid  = threadIdx.x;
    const int lane = tid & 31;
    const int wav  = tid >> 5;           // 0..7
    const int lr   = lane & 15;          // row-in-tile
    const int lh   = lane >> 4;          // lane half
    const int wm   = wav >> 2;           // 0..1  (token sub-block)
    const int wn   = wav & 3;            // 0..3  (feature sub-block)

    const int tBlk = blockIdx.y * 128;   // block token base
    const int nBlk = blockIdx.x * 256;   // block feature base

    // ---- async tile loader: A = 512 x b128 chunks, B = 1024 x b128 chunks
    auto load_stage = [&](int buf, int k0) {
#pragma unroll
        for (int q = 0; q < 2; ++q) {                 // A: 2 chunks / thread
            int c = tid + 256 * q;
            int row = c >> 2, ch = c & 3;
            async_copy_b128(xh + (long)(tBlk + row) * MDIM + k0 + ch * 8,
                            &As[buf * ABUF + row * RS + ch * 8]);
        }
#pragma unroll
        for (int q = 0; q < 4; ++q) {                 // B: 4 chunks / thread
            int c = tid + 256 * q;
            int row = c >> 2, ch = c & 3;
            async_copy_b128(wd + (long)(nBlk + row) * MDIM + k0 + ch * 8,
                            &Bs[buf * BBUF + row * RS + ch * 8]);
        }
    };

    // ---- per-lane LDS fragment pointers (ISA 7.12.2 operand layouts)
    // A (16x32): lanes 0-15 hold K{0..7,16..23}; lanes 16-31 K{8..15,24..31}
    // B (32x16): lanes 0-15 hold K{0..7,16..23}; lanes 16-31 K{8..15,24..31}
    const _Float16* aF[4];
    const _Float16* bF[4];
#pragma unroll
    for (int i = 0; i < 4; ++i)
        aF[i] = &As[(wm * 64 + i * 16 + lr) * RS + lh * 8];
#pragma unroll
    for (int j = 0; j < 4; ++j)
        bF[j] = &Bs[(wn * 64 + j * 16 + lr) * RS + lh * 8];

    v8f acc[4][4];
#pragma unroll
    for (int i = 0; i < 4; ++i)
#pragma unroll
        for (int j = 0; j < 4; ++j)
            acc[i][j] = (v8f){0.f,0.f,0.f,0.f,0.f,0.f,0.f,0.f};

    union Frag { v16h v; v8h h[2]; };

    constexpr int NSTEP = MDIM / 32;     // 128 K-steps

    // prologue: stage 0
    load_stage(0, 0);
    wait_asynccnt0();
    __syncthreads();

    for (int it = 0; it < NSTEP; ++it) {
        const int cur = it & 1;
        if (it + 1 < NSTEP)
            load_stage(cur ^ 1, (it + 1) * 32);      // DMA next tile

        const int ao = cur * ABUF;
        const int bo = cur * BBUF;

        Frag a[4], b[4];
#pragma unroll
        for (int i = 0; i < 4; ++i) {
            a[i].h[0] = *(const v8h*)(aF[i] + ao);        // K lo run
            a[i].h[1] = *(const v8h*)(aF[i] + ao + 16);   // K hi run
        }
#pragma unroll
        for (int j = 0; j < 4; ++j) {
            b[j].h[0] = *(const v8h*)(bF[j] + bo);        // K lo run
            b[j].h[1] = *(const v8h*)(bF[j] + bo + 16);   // K hi run
        }
#pragma unroll
        for (int i = 0; i < 4; ++i)
#pragma unroll
            for (int j = 0; j < 4; ++j)
                acc[i][j] = __builtin_amdgcn_wmma_f32_16x16x32_f16(
                    false, a[i].v, false, b[j].v,
                    (short)0, acc[i][j], false, false);

        wait_asynccnt0();    // next stage resident in LDS
        __syncthreads();     // all waves done reading cur stage
    }

    // ---- epilogue: C/D layout lane->N = lane%16, VGPR r -> M = r + 8*(lane/16)
#pragma unroll
    for (int j = 0; j < 4; ++j) {
        const int n = nBlk + wn * 64 + j * 16 + lr;
        const float bv = bias[n];
#pragma unroll
        for (int i = 0; i < 4; ++i) {
#pragma unroll
            for (int r = 0; r < 8; ++r) {
                const int t = tBlk + wm * 64 + i * 16 + r + 8 * lh;
                out[(long)t * NDIM + n] = acc[i][j][r] + bv;
            }
        }
    }
}

// ---------------------------------------------------------------------------
// launcher
// ---------------------------------------------------------------------------
extern "C" void kernel_launch(void* const* d_in, const int* in_sizes, int n_in,
                              void* d_out, int out_size, void* d_ws, size_t ws_size,
                              hipStream_t stream)
{
    const float* x    = (const float*)d_in[0];
    const float* wval = (const float*)d_in[1];
    const float* bias = (const float*)d_in[2];
    const int*   rows = (const int*)d_in[3];
    const int*   cols = (const int*)d_in[4];
    const int    nnz  = in_sizes[1];

    // workspace: [ xh : BATCH*MDIM f16 ][ wd : NDIM*MDIM f16 ]  (~64 MB)
    _Float16* xh = (_Float16*)d_ws;
    _Float16* wd = xh + (size_t)BATCH * MDIM;

    cvt_f32_to_f16<<<4096, 256, 0, stream>>>(x, xh, (long)BATCH * MDIM);
    zero_f16<<<4096, 256, 0, stream>>>(wd, (long)NDIM * MDIM);
    scatter_csr<<<(nnz + 255) / 256, 256, 0, stream>>>(wval, rows, cols, wd, nnz);

    dim3 grid(NDIM / 256, BATCH / 128);   // 16 x 32 blocks, 8 waves each
    gemm_wmma_f16<<<grid, 256, 0, stream>>>(xh, wd, bias, (float*)d_out);
}
